// TransNetNSV3_83133386981996
// MI455X (gfx1250) — matrix operand order repaired and checked by reading, not compiled, not run on hardware
//
#include <hip/hip_runtime.h>
#include <hip/hip_bf16.h>
#include <math.h>

typedef __attribute__((ext_vector_type(16))) _Float16 v16h;
typedef __attribute__((ext_vector_type(8)))  float    v8f;

#define N0_  80000
#define N1_  40000
#define N2_  20000
#define E1_  500000
#define E2_  250000
#define NTYPE_ 23
#define NTS_   1158

static inline int cdiv(long a, long b) { return (int)((a + b - 1) / b); }

// ---------------- order-preserving float<->uint encoding for atomicMax ----
__device__ inline unsigned enc_f32(float f) {
    unsigned u = __float_as_uint(f);
    return (u & 0x80000000u) ? ~u : (u | 0x80000000u);
}
__device__ inline float dec_f32(unsigned u) {
    unsigned b = (u & 0x80000000u) ? (u ^ 0x80000000u) : ~u;
    return __uint_as_float(b);
}

// ---------------- wave32 reductions ----------------
__device__ inline float wave_sum(float x) {
    for (int o = 16; o; o >>= 1) x += __shfl_xor(x, o, 32);
    return x;
}
__device__ inline float wave_max(float x) {
    for (int o = 16; o; o >>= 1) x = fmaxf(x, __shfl_xor(x, o, 32));
    return x;
}

// =================================================================
// WMMA GEMM: out[M,N] = A[M,K] @ W[N,K]^T + bias[N]
// f32 inputs converted to f16 in LDS, f32 accumulate via
// v_wmma_f32_16x16x32_f16.  256 threads = 8 waves; each wave owns a
// 16x16 output tile (wave w -> columns [16w,16w+16)).  Whole W is
// staged in LDS (<=128x128 f16 = 32KB).
// =================================================================
__global__ __launch_bounds__(256)
void gemm_xWT(const float* __restrict__ A, const float* __restrict__ W,
              const float* __restrict__ bias, float* __restrict__ out,
              int M, int N, int K) {
    __shared__ _Float16 Wl[128 * 128];
    __shared__ _Float16 Al[16 * 128];
    const int tid = threadIdx.x;
    const int mblk = blockIdx.x * 16;

    for (int i = tid; i < N * K; i += 256) Wl[i] = (_Float16)W[i];
    for (int i = tid; i < 16 * K; i += 256) {
        int r = i / K, c = i - r * K;
        int gr = mblk + r;
        Al[i] = (gr < M) ? (_Float16)A[(long)gr * K + c] : (_Float16)0.0f;
    }
    __syncthreads();

    const int wave = tid >> 5, lane = tid & 31;
    const int nbase = wave * 16;
    if (nbase >= N) return;                 // wave-uniform branch (EXEC stays full)
    const int half = lane >> 4;             // 0: lanes 0-15, 1: lanes 16-31
    const int lrow = lane & 15;

    v8f acc = {};
    for (int kk = 0; kk < K; kk += 32) {
        v16h a, b;
        // A fragment 16x32 (ISA 7.12.2): lane L holds row M=L&15;
        // VGPR v<4 -> K = half*8 + 2v, VGPR v>=4 -> K = 16 + half*8 + 2(v-4)
        const _Float16* ar = &Al[lrow * K + kk];
        #pragma unroll
        for (int v = 0; v < 4; ++v) {
            int k0 = half * 8 + 2 * v;
            a[2 * v]     = ar[k0];
            a[2 * v + 1] = ar[k0 + 1];
            int k1 = 16 + half * 8 + 2 * v;
            a[2 * v + 8] = ar[k1];
            a[2 * v + 9] = ar[k1 + 1];
        }
        // B fragment 32x16: B[k][n] = W[nbase+n][k]; lane holds col n=L&15,
        // same K layout as A (K = half*8 + 2v, 16 + half*8 + 2(v-4))
        const _Float16* br = &Wl[(nbase + lrow) * K + kk];
        #pragma unroll
        for (int v = 0; v < 4; ++v) {
            int k0 = half * 8 + 2 * v;
            b[2 * v]     = br[k0];
            b[2 * v + 1] = br[k0 + 1];
            int k1 = 16 + half * 8 + 2 * v;
            b[2 * v + 8] = br[k1];
            b[2 * v + 9] = br[k1 + 1];
        }
        acc = __builtin_amdgcn_wmma_f32_16x16x32_f16(
            false, a, false, b, (short)0, acc, false, false);
    }
    // D layout: VGPR r -> (M = r + 8*half, N = lane&15)
    #pragma unroll
    for (int r = 0; r < 8; ++r) {
        int m = mblk + r + half * 8;
        int n = nbase + lrow;
        if (m < M) out[(long)m * N + n] = acc[r] + bias[n];
    }
}

// =================================================================
// Edge-feature tables: e_edge = TA[et] + TB[ets]
//   TA[t][n]  = sum_d edge_W[d][t]*We[n][d] + (bias terms folded in)
//   TB[ts][n] = sum_d time_W[d][ts]*We[n][10+d]
// =================================================================
__global__ void build_tableA(const float* __restrict__ edge_W,
                             const float* __restrict__ edge_b,
                             const float* __restrict__ time_b,
                             const float* __restrict__ We,
                             float* __restrict__ TA, int NT, int N) {
    int i = blockIdx.x * blockDim.x + threadIdx.x;
    if (i >= NT * N) return;
    int t = i / N, n = i - t * N;
    float acc = 0.f;
    for (int d = 0; d < 10; ++d) {
        acc += (edge_W[d * NT + t] + edge_b[d]) * We[n * 20 + d];
        acc += time_b[d] * We[n * 20 + 10 + d];
    }
    TA[i] = acc;
}
__global__ void build_tableB(const float* __restrict__ time_W,
                             const float* __restrict__ We,
                             float* __restrict__ TB, int NTS, int N) {
    int i = blockIdx.x * blockDim.x + threadIdx.x;
    if (i >= NTS * N) return;
    int t = i / N, n = i - t * N;
    float acc = 0.f;
    for (int d = 0; d < 10; ++d)
        acc += time_W[d * NTS + t] * We[n * 20 + 10 + d];
    TB[i] = acc;
}

// =================================================================
// Edge pass 1: alpha[e][h] = <q[dst], k[src]+e> / 8; segment max via
// encoded-uint atomicMax.  8 lanes per edge (4 per head).
// =================================================================
__global__ void edge_pass1(const int* __restrict__ src, const int* __restrict__ dst,
                           const int* __restrict__ et, const int* __restrict__ ets,
                           const float* __restrict__ q, const float* __restrict__ k,
                           const float* __restrict__ TA, const float* __restrict__ TB,
                           float* __restrict__ alpha, unsigned* __restrict__ menc,
                           int E, float scale) {
    long t = (long)blockIdx.x * blockDim.x + threadIdx.x;
    int e = (int)(t >> 3);
    if (e >= E) return;
    int lg = (int)(t & 7);
    int s = src[e], d = dst[e], te = et[e], ts = ets[e];
    int h = lg >> 2, seg = lg & 3;
    int base = h * 64 + seg * 16;
    const float* qd = q  + (long)d  * 128 + base;
    const float* ks = k  + (long)s  * 128 + base;
    const float* ta = TA + (long)te * 128 + base;
    const float* tb = TB + (long)ts * 128 + base;
    float acc = 0.f;
    #pragma unroll
    for (int i = 0; i < 16; ++i) acc += qd[i] * (ks[i] + ta[i] + tb[i]);
    acc += __shfl_xor(acc, 1, 32);
    acc += __shfl_xor(acc, 2, 32);
    if (seg == 0) {
        float al = acc * scale;
        alpha[(long)e * 2 + h] = al;
        atomicMax(&menc[(long)d * 2 + h], enc_f32(al));
    }
}

// =================================================================
// Edge pass 2: ex = exp(alpha - m[dst]); accumulate unnormalized
//   s[dst][h]   += ex
//   agg[dst][f] += ex * (v[src][f] + e[f])
// One wave (32 lanes) per edge; 4 features per lane.
// =================================================================
__global__ void edge_pass2(const int* __restrict__ src, const int* __restrict__ dst,
                           const int* __restrict__ et, const int* __restrict__ ets,
                           const float* __restrict__ v,
                           const float* __restrict__ TA, const float* __restrict__ TB,
                           const float* __restrict__ alpha,
                           const unsigned* __restrict__ menc,
                           float* __restrict__ sacc, float* __restrict__ agg, int E) {
    long t = (long)blockIdx.x * blockDim.x + threadIdx.x;
    int e = (int)(t >> 5);
    if (e >= E) return;
    int lane = (int)(t & 31);
    int s = src[e], d = dst[e], te = et[e], ts = ets[e];
    int f0 = lane * 4;
    int h = lane >> 4;                       // feature head
    float m = dec_f32(menc[(long)d * 2 + h]);
    float ex = __expf(alpha[(long)e * 2 + h] - m);
    if (lane == 0 || lane == 16) atomicAdd(&sacc[(long)d * 2 + h], ex);
    const float* vs = v  + (long)s  * 128 + f0;
    const float* ta = TA + (long)te * 128 + f0;
    const float* tb = TB + (long)ts * 128 + f0;
    float* ag = agg + (long)d * 128 + f0;
    #pragma unroll
    for (int j = 0; j < 4; ++j)
        atomicAdd(&ag[j], ex * (vs[j] + ta[j] + tb[j]));
}

// =================================================================
// Layer-1 node epilogue: normalize, beta-gate with skip, layernorm,
// exact gelu.  One wave per node (8 nodes / 256-thread block).
// =================================================================
__global__ __launch_bounds__(256)
void node_fin1(const float* __restrict__ agg, const float* __restrict__ sacc,
               const float* __restrict__ skip, const float* __restrict__ Wb,
               const float* __restrict__ ln_g, const float* __restrict__ ln_b,
               float* __restrict__ hout, int Nd) {
    int n = blockIdx.x * 8 + (threadIdx.x >> 5);
    if (n >= Nd) return;
    int lane = threadIdx.x & 31;
    int h = lane >> 4;
    float sh = sacc[(long)n * 2 + h] + 1e-16f;
    float o[4], xr[4], bp = 0.f;
    #pragma unroll
    for (int j = 0; j < 4; ++j) {
        int f = 4 * lane + j;
        o[j]  = agg[(long)n * 128 + f] / sh;
        xr[j] = skip[(long)n * 128 + f];
        bp += Wb[f] * o[j] + Wb[128 + f] * xr[j] + Wb[256 + f] * (o[j] - xr[j]);
    }
    float beta = 1.f / (1.f + __expf(-wave_sum(bp)));
    float g[4], msum = 0.f, vsum = 0.f;
    #pragma unroll
    for (int j = 0; j < 4; ++j) {
        g[j] = beta * xr[j] + (1.f - beta) * o[j];
        msum += g[j];
        vsum += g[j] * g[j];
    }
    float mu  = wave_sum(msum) * (1.f / 128.f);
    float var = wave_sum(vsum) * (1.f / 128.f) - mu * mu;
    float rstd = rsqrtf(var + 1e-5f);
    #pragma unroll
    for (int j = 0; j < 4; ++j) {
        int f = 4 * lane + j;
        float y = (g[j] - mu) * rstd * ln_g[f] + ln_b[f];
        hout[(long)n * 128 + f] = 0.5f * y * (1.f + erff(y * 0.70710678f));
    }
}

// =================================================================
// Layer-2 node epilogue: normalize per head, mean over heads,
// beta-gate, log_softmax over 64.  One wave per node, 2 feats/lane.
// =================================================================
__global__ __launch_bounds__(256)
void node_fin2(const float* __restrict__ agg, const float* __restrict__ sacc,
               const float* __restrict__ skip, const float* __restrict__ Wb,
               float* __restrict__ out, int Nd) {
    int n = blockIdx.x * 8 + (threadIdx.x >> 5);
    if (n >= Nd) return;
    int lane = threadIdx.x & 31;
    float s0 = sacc[(long)n * 2 + 0] + 1e-16f;
    float s1 = sacc[(long)n * 2 + 1] + 1e-16f;
    float r[2], bp = 0.f;
    #pragma unroll
    for (int j = 0; j < 2; ++j) {
        int c = lane + 32 * j;
        float o = 0.5f * (agg[(long)n * 128 + c]      / s0 +
                          agg[(long)n * 128 + 64 + c] / s1);
        float xr = skip[(long)n * 64 + c];
        bp += Wb[c] * o + Wb[64 + c] * xr + Wb[128 + c] * (o - xr);
        r[j] = o;           // temporarily store o; gate after reduction
        // stash xr in high part via recompute below (cheap reload)
    }
    float beta = 1.f / (1.f + __expf(-wave_sum(bp)));
    #pragma unroll
    for (int j = 0; j < 2; ++j) {
        int c = lane + 32 * j;
        float xr = skip[(long)n * 64 + c];
        r[j] = beta * xr + (1.f - beta) * r[j];
    }
    float mx = wave_max(fmaxf(r[0], r[1]));
    float es = wave_sum(__expf(r[0] - mx) + __expf(r[1] - mx));
    float lse = logf(es);
    #pragma unroll
    for (int j = 0; j < 2; ++j) {
        int c = lane + 32 * j;
        out[(long)n * 64 + c] = r[j] - mx - lse;
    }
}

// =================================================================
extern "C" void kernel_launch(void* const* d_in, const int* in_sizes, int n_in,
                              void* d_out, int out_size, void* d_ws, size_t ws_size,
                              hipStream_t stream) {
    const float* x      = (const float*)d_in[0];
    const int* ei1_src  = (const int*)d_in[1];
    const int* ei1_dst  = (const int*)d_in[2];
    const int* et1      = (const int*)d_in[3];
    const int* ets1     = (const int*)d_in[4];
    const int* ei2_src  = (const int*)d_in[5];
    const int* ei2_dst  = (const int*)d_in[6];
    const int* et2      = (const int*)d_in[7];
    const int* ets2     = (const int*)d_in[8];
    const float* edge_W = (const float*)d_in[9];
    const float* edge_b = (const float*)d_in[10];
    const float* time_W = (const float*)d_in[11];
    const float* time_b = (const float*)d_in[12];
    const float* Wq1 = (const float*)d_in[13]; const float* bq1 = (const float*)d_in[14];
    const float* Wk1 = (const float*)d_in[15]; const float* bk1 = (const float*)d_in[16];
    const float* Wv1 = (const float*)d_in[17]; const float* bv1 = (const float*)d_in[18];
    const float* We1 = (const float*)d_in[19];
    const float* Wskip1 = (const float*)d_in[20]; const float* bskip1 = (const float*)d_in[21];
    const float* Wbeta1 = (const float*)d_in[22];
    const float* ln_g = (const float*)d_in[23]; const float* ln_b = (const float*)d_in[24];
    const float* Wq2 = (const float*)d_in[25]; const float* bq2 = (const float*)d_in[26];
    const float* Wk2 = (const float*)d_in[27]; const float* bk2 = (const float*)d_in[28];
    const float* Wv2 = (const float*)d_in[29]; const float* bv2 = (const float*)d_in[30];
    const float* We2 = (const float*)d_in[31];
    const float* Wskip2 = (const float*)d_in[32]; const float* bskip2 = (const float*)d_in[33];
    const float* Wbeta2 = (const float*)d_in[34];

    // ---- workspace layout (floats) ----
    float* ws = (float*)d_ws;
    size_t off = 0;
    float* k1    = ws + off; off += (size_t)N0_ * 128;
    float* v1    = ws + off; off += (size_t)N0_ * 128;
    float* q1    = ws + off; off += (size_t)N1_ * 128;
    float* skip1 = ws + off; off += (size_t)N1_ * 128;
    float* TA1   = ws + off; off += (size_t)NTYPE_ * 128;
    float* TB1   = ws + off; off += (size_t)NTS_ * 128;
    float* al1   = ws + off; off += (size_t)E1_ * 2;
    unsigned* m1 = (unsigned*)(ws + off); off += (size_t)N1_ * 2;
    float* s1    = ws + off; off += (size_t)N1_ * 2;
    float* agg1  = ws + off; off += (size_t)N1_ * 128;
    float* hbuf  = ws + off; off += (size_t)N1_ * 128;
    float* k2    = ws + off; off += (size_t)N1_ * 128;
    float* v2    = ws + off; off += (size_t)N1_ * 128;
    float* q2    = ws + off; off += (size_t)N2_ * 128;
    float* skip2 = ws + off; off += (size_t)N2_ * 64;
    float* TA2   = ws + off; off += (size_t)NTYPE_ * 128;
    float* TB2   = ws + off; off += (size_t)NTS_ * 128;
    float* al2   = ws + off; off += (size_t)E2_ * 2;
    unsigned* m2 = (unsigned*)(ws + off); off += (size_t)N2_ * 2;
    float* s2    = ws + off; off += (size_t)N2_ * 2;
    float* agg2  = ws + off; off += (size_t)N2_ * 128;

    const float scale = 0.125f;   // 1/sqrt(64)

    // ---- edge tables (biases folded into TA) ----
    build_tableA<<<cdiv(NTYPE_ * 128, 256), 256, 0, stream>>>(edge_W, edge_b, time_b, We1, TA1, NTYPE_, 128);
    build_tableB<<<cdiv((long)NTS_ * 128, 256), 256, 0, stream>>>(time_W, We1, TB1, NTS_, 128);
    build_tableA<<<cdiv(NTYPE_ * 128, 256), 256, 0, stream>>>(edge_W, edge_b, time_b, We2, TA2, NTYPE_, 128);
    build_tableB<<<cdiv((long)NTS_ * 128, 256), 256, 0, stream>>>(time_W, We2, TB2, NTS_, 128);

    // ---- layer 1 dense projections (WMMA) ----
    gemm_xWT<<<cdiv(N0_, 16), 256, 0, stream>>>(x, Wk1, bk1, k1, N0_, 128, 128);
    gemm_xWT<<<cdiv(N0_, 16), 256, 0, stream>>>(x, Wv1, bv1, v1, N0_, 128, 128);
    gemm_xWT<<<cdiv(N1_, 16), 256, 0, stream>>>(x, Wq1, bq1, q1, N1_, 128, 128);
    gemm_xWT<<<cdiv(N1_, 16), 256, 0, stream>>>(x, Wskip1, bskip1, skip1, N1_, 128, 128);

    hipMemsetAsync(m1, 0, (size_t)N1_ * 2 * sizeof(unsigned), stream);
    hipMemsetAsync(s1, 0, (size_t)N1_ * 2 * sizeof(float), stream);
    hipMemsetAsync(agg1, 0, (size_t)N1_ * 128 * sizeof(float), stream);

    edge_pass1<<<cdiv((long)E1_ * 8, 256), 256, 0, stream>>>(
        ei1_src, ei1_dst, et1, ets1, q1, k1, TA1, TB1, al1, m1, E1_, scale);
    edge_pass2<<<cdiv((long)E1_ * 32, 256), 256, 0, stream>>>(
        ei1_src, ei1_dst, et1, ets1, v1, TA1, TB1, al1, m1, s1, agg1, E1_);

    node_fin1<<<cdiv(N1_, 8), 256, 0, stream>>>(agg1, s1, skip1, Wbeta1, ln_g, ln_b, hbuf, N1_);

    // ---- layer 2 dense projections (WMMA) ----
    gemm_xWT<<<cdiv(N1_, 16), 256, 0, stream>>>(hbuf, Wk2, bk2, k2, N1_, 128, 128);
    gemm_xWT<<<cdiv(N1_, 16), 256, 0, stream>>>(hbuf, Wv2, bv2, v2, N1_, 128, 128);
    gemm_xWT<<<cdiv(N2_, 16), 256, 0, stream>>>(hbuf, Wq2, bq2, q2, N2_, 128, 128);
    gemm_xWT<<<cdiv(N2_, 16), 256, 0, stream>>>(hbuf, Wskip2, bskip2, skip2, N2_, 64, 128);

    hipMemsetAsync(m2, 0, (size_t)N2_ * 2 * sizeof(unsigned), stream);
    hipMemsetAsync(s2, 0, (size_t)N2_ * 2 * sizeof(float), stream);
    hipMemsetAsync(agg2, 0, (size_t)N2_ * 128 * sizeof(float), stream);

    edge_pass1<<<cdiv((long)E2_ * 8, 256), 256, 0, stream>>>(
        ei2_src, ei2_dst, et2, ets2, q2, k2, TA2, TB2, al2, m2, E2_, scale);
    edge_pass2<<<cdiv((long)E2_ * 32, 256), 256, 0, stream>>>(
        ei2_src, ei2_dst, et2, ets2, v2, TA2, TB2, al2, m2, s2, agg2, E2_);

    node_fin2<<<cdiv(N2_, 8), 256, 0, stream>>>(agg2, s2, skip2, Wbeta2, (float*)d_out, N2_);
}
